// RBFKernelFFN_61186104098896
// MI455X (gfx1250) — matrix operand order repaired and checked by reading, not compiled, not run on hardware
//
#include <hip/hip_runtime.h>
#include <hip/hip_bf16.h>

// ---------------------------------------------------------------------------
// RBF Kernel FFN, fused bf16-WMMA implementation for gfx1250 (MI455X).
//   out = exp(-||x-c||^2 / 2) @ w_out^T + b_out
// GEMM1 (x . centers^T) and GEMM2 (rbf . w_out^T) both run on
// v_wmma_f32_16x16x32_bf16; the rbf intermediate stays in LDS.
// ---------------------------------------------------------------------------

typedef __attribute__((ext_vector_type(16))) __bf16 v16bf;
typedef __attribute__((ext_vector_type(8)))  __bf16 v8bf;
typedef __attribute__((ext_vector_type(8)))  float  v8f;

#define EMB    768
#define NKER   3072
#define ROWS   16384      // 4 * 4096
#define BK     128        // K-chunk per iteration (8 waves * 16)
#define NCHUNK (NKER / BK)

static __device__ __forceinline__ v16bf cat16(v8bf lo, v8bf hi) {
    return __builtin_shufflevector(lo, hi, 0, 1, 2, 3, 4, 5, 6, 7,
                                           8, 9, 10, 11, 12, 13, 14, 15);
}

// ---- prep: centers f32 -> bf16, plus ||c||^2 per row ----------------------
__global__ __launch_bounds__(256) void prep_centers(const float* __restrict__ centers,
                                                    __hip_bfloat16* __restrict__ c_bf,
                                                    float* __restrict__ c_sq) {
    __shared__ float red[256];
    const int row = blockIdx.x;          // 0..3071
    const int t   = threadIdx.x;         // 0..255
    float s = 0.f;
    const float* src = centers + (size_t)row * EMB;
    __hip_bfloat16* dst = c_bf + (size_t)row * EMB;
    #pragma unroll
    for (int i = t; i < EMB; i += 256) {
        float v = src[i];
        s += v * v;
        dst[i] = __float2bfloat16(v);
    }
    red[t] = s;
    __syncthreads();
    #pragma unroll
    for (int off = 128; off > 0; off >>= 1) {
        if (t < off) red[t] += red[t + off];
        __syncthreads();
    }
    if (t == 0) c_sq[row] = red[0];
}

// ---- prep: w_out f32 -> bf16 (same (D,K) row-major layout) ----------------
__global__ __launch_bounds__(256) void prep_w(const float* __restrict__ w,
                                              __hip_bfloat16* __restrict__ w_bf,
                                              int n) {
    int i = blockIdx.x * 256 + threadIdx.x;
    if (i < n) w_bf[i] = __float2bfloat16(w[i]);
}

// ---- fused main kernel ----------------------------------------------------
__global__ __launch_bounds__(256) void rbf_ffn(const float* __restrict__ x,
                                               const __hip_bfloat16* __restrict__ c_bf,
                                               const float* __restrict__ c_sq,
                                               const __hip_bfloat16* __restrict__ w_bf,
                                               const float* __restrict__ b_out,
                                               float* __restrict__ out) {
    __shared__ __hip_bfloat16 x_lds[16][EMB];   // 24 KB: 16-row x tile, bf16
    __shared__ __hip_bfloat16 rbf_lds[16][BK];  // 4 KB : rbf chunk, bf16
    __shared__ float red[16][16];
    __shared__ float xsq[16];

    const int t    = threadIdx.x;
    const int lane = t & 31;
    const int wv   = t >> 5;         // 0..7
    const int hi   = (lane >= 16) ? 1 : 0;
    const int nloc = lane & 15;
    const int m0   = blockIdx.x * 16;

    // ---- stage x tile (f32 -> bf16) + per-row sum of squares --------------
    {
        const int r  = t >> 4;             // 0..15
        const int cb = (t & 15) * 48;      // 16 threads * 48 = 768
        const float* xr = x + (size_t)(m0 + r) * EMB + cb;
        float s = 0.f;
        #pragma unroll 12
        for (int i = 0; i < 48; ++i) {
            float v = xr[i];
            s += v * v;
            x_lds[r][cb + i] = __float2bfloat16(v);
        }
        red[r][t & 15] = s;
    }
    __syncthreads();
    if (t < 16) {
        float s = 0.f;
        #pragma unroll
        for (int i = 0; i < 16; ++i) s += red[t][i];
        xsq[t] = s;
    }
    __syncthreads();

    // per-lane copies of ||x||^2 in C/D layout (VGPR r -> row r + hi*8)
    float xs[8];
    #pragma unroll
    for (int r = 0; r < 8; ++r) xs[r] = xsq[r + hi * 8];

    // output accumulators: 6 N-tiles of 16 cols per wave (8*6*16 = 768)
    v8f acc2[6];
    #pragma unroll
    for (int i = 0; i < 6; ++i) acc2[i] = (v8f){0.f,0.f,0.f,0.f,0.f,0.f,0.f,0.f};

    for (int chunk = 0; chunk < NCHUNK; ++chunk) {
        const int k0 = chunk * BK;
        const int kc = k0 + wv * 16 + nloc;      // this lane's center row (GEMM1 B col)
        const float csq = c_sq[kc];

        // ---- GEMM1: acc1 = x_tile(16x768) . centers^T chunk (768x16) ------
        v8f acc1 = (v8f){0.f,0.f,0.f,0.f,0.f,0.f,0.f,0.f};
        const __hip_bfloat16* crow = c_bf + (size_t)kc * EMB;
        #pragma unroll 4
        for (int d0 = 0; d0 < EMB; d0 += 32) {
            // A fragment (16x32 bf16) from LDS:
            //   elems 0..15 -> K = d0 + hi*16 .. +15 (contiguous)
            const v8bf* pa = (const v8bf*)&x_lds[nloc][d0 + hi * 16];
            v16bf a = cat16(pa[0], pa[1]);
            // B fragment (32x16 bf16) from global bf16 centers:
            //   lane = column, elems -> K = d0 + hi*16 .. +15 (contiguous)
            const v8bf* pb = (const v8bf*)(crow + d0 + hi * 16);
            v16bf b = cat16(pb[0], pb[1]);
            acc1 = __builtin_amdgcn_wmma_f32_16x16x32_bf16(
                       false, a, false, b, (short)0, acc1, false, false);
        }

        // ---- rbf = exp(-(||x||^2 - 2 x.c + ||c||^2)/2) -> LDS bf16 --------
        #pragma unroll
        for (int r = 0; r < 8; ++r) {
            float dist = xs[r] - 2.f * acc1[r] + csq;
            float v = __expf(-0.5f * dist);
            rbf_lds[r + hi * 8][wv * 16 + nloc] = __float2bfloat16(v);
        }
        __syncthreads();

        // ---- GEMM2: out(16x768) += rbf(16x128) . w_out^T chunk (128x768) --
        v16bf a2[4];
        #pragma unroll
        for (int kk = 0; kk < 4; ++kk) {
            const v8bf* pa = (const v8bf*)&rbf_lds[nloc][kk * 32 + hi * 16];
            a2[kk] = cat16(pa[0], pa[1]);
        }
        #pragma unroll
        for (int nt = 0; nt < 6; ++nt) {
            const int n0 = wv * 96 + nt * 16;
            // B fragment: lane = out column (n0+nloc), elems -> K contiguous
            const __hip_bfloat16* wrow =
                w_bf + (size_t)(n0 + nloc) * NKER + k0 + hi * 16;
            #pragma unroll
            for (int kk = 0; kk < 4; ++kk) {
                const v8bf* pb = (const v8bf*)(wrow + kk * 32);
                v16bf b = cat16(pb[0], pb[1]);
                acc2[nt] = __builtin_amdgcn_wmma_f32_16x16x32_bf16(
                               false, a2[kk], false, b, (short)0, acc2[nt],
                               false, false);
            }
        }
        __syncthreads();   // protect rbf_lds before next chunk overwrites it
    }

    // ---- epilogue: add bias, store f32 ------------------------------------
    #pragma unroll
    for (int nt = 0; nt < 6; ++nt) {
        const int n = wv * 96 + nt * 16 + nloc;
        const float bias = b_out[n];
        #pragma unroll
        for (int r = 0; r < 8; ++r) {
            out[(size_t)(m0 + r + hi * 8) * EMB + n] = acc2[nt][r] + bias;
        }
    }
}

// ---------------------------------------------------------------------------
extern "C" void kernel_launch(void* const* d_in, const int* in_sizes, int n_in,
                              void* d_out, int out_size, void* d_ws, size_t ws_size,
                              hipStream_t stream) {
    const float* x       = (const float*)d_in[0];   // (4,4096,768)
    const float* centers = (const float*)d_in[1];   // (3072,768)
    const float* w_out   = (const float*)d_in[2];   // (768,3072)
    const float* b_out   = (const float*)d_in[3];   // (768,)
    float* out = (float*)d_out;

    char* ws = (char*)d_ws;
    const size_t c_bytes = (size_t)NKER * EMB * sizeof(__hip_bfloat16);
    __hip_bfloat16* c_bf = (__hip_bfloat16*)ws;
    __hip_bfloat16* w_bf = (__hip_bfloat16*)(ws + c_bytes);
    float*          csq  = (float*)(ws + 2 * c_bytes);

    prep_centers<<<NKER, 256, 0, stream>>>(centers, c_bf, csq);
    prep_w<<<(EMB * NKER + 255) / 256, 256, 0, stream>>>(w_out, w_bf, EMB * NKER);
    rbf_ffn<<<ROWS / 16, 256, 0, stream>>>(x, c_bf, csq, w_bf, b_out, out);
}
